// TreeGCN_48911087567501
// MI455X (gfx1250) — compile-verified, matrix operand order checked by reading, not run on hardware
//
#include <hip/hip_runtime.h>
#include <math.h>

#define NN    40000
#define TSEQ  30
#define DIN   100
#define HD    100
#define NE    80000
#define BATCH 128
#define KP    128   // padded K for 100-dim reductions (4 chunks of 32)
#define KP2   256   // padded K for 200-dim reduction (8 chunks of 32)
#define NP3   304   // padded 3H (19 WMMA N-tiles)
#define NT3   19
#define NP1   112   // padded H (7 WMMA N-tiles)
#define NT1   7
#define MBLK  32    // nodes per block (2 WMMA M-tiles)
#define MTILES 2

typedef _Float16 v16h __attribute__((ext_vector_type(16)));
typedef float    v8f  __attribute__((ext_vector_type(8)));

__device__ __forceinline__ v8f wmma_f16(v16h a, v16h b, v8f c) {
  // D = A(16x32 f16) x B(32x16 f16) + C(16x16 f32)
  return __builtin_amdgcn_wmma_f32_16x16x32_f16(false, a, false, b, (short)0, c, false, false);
}

__device__ __forceinline__ float sigmoidf_(float x) { return 1.0f / (1.0f + expf(-x)); }
__device__ __forceinline__ float eluf_(float x)     { return x > 0.0f ? x : (expf(x) - 1.0f); }

// ---------------------------------------------------------------------------
// GRU tile matmul: gi = A_x @ Wih^T , gh = A_h @ Whh^T  (K=128 padded, N=304)
// A tiles in LDS (f16, [MBLK][KP]); weights global f16 [NP3][KP].
// 8 waves split 2*19 = 38 (matrix, N-tile) units; B fragments are loaded ONCE
// per unit into registers and reused across the 2 M-tiles (L2 traffic /2 vs
// the 16-node version).
// ---------------------------------------------------------------------------
__device__ __forceinline__ void matmul_phase(const _Float16* xA, const _Float16* hA,
                                             const _Float16* wih, const _Float16* whh,
                                             float* gi, float* gh, int wave, int lane)
{
  for (int u = wave; u < 2 * NT3; u += 8) {
    const _Float16* Als = (u < NT3) ? xA : hA;
    const _Float16* Wg  = (u < NT3) ? wih : whh;
    float*          Ols = (u < NT3) ? gi  : gh;
    const int nt = (u < NT3) ? u : (u - NT3);

    const _Float16* bp = Wg + (size_t)(nt * 16 + (lane & 15)) * KP + ((lane >> 4) << 4);
    v16h b0 = *(const v16h*)(bp);
    v16h b1 = *(const v16h*)(bp + 32);
    v16h b2 = *(const v16h*)(bp + 64);
    v16h b3 = *(const v16h*)(bp + 96);

    const int row0 = (lane >> 4) << 3;
    const int col  = nt * 16 + (lane & 15);
#pragma unroll
    for (int mt = 0; mt < MTILES; ++mt) {
      const _Float16* ap = Als + (size_t)(mt * 16 + (lane & 15)) * KP + ((lane >> 4) << 4);
      v8f c = {};
      c = wmma_f16(*(const v16h*)(ap),      b0, c);
      c = wmma_f16(*(const v16h*)(ap + 32), b1, c);
      c = wmma_f16(*(const v16h*)(ap + 64), b2, c);
      c = wmma_f16(*(const v16h*)(ap + 96), b3, c);
#pragma unroll
      for (int i = 0; i < 8; ++i)
        Ols[(mt * 16 + row0 + i) * NP3 + col] = c[i];
    }
  }
}

// GRU gate math (torch order r,z,n). Updates hold in place; optionally emits
// f16 copy of h_new (zero padded) as the next layer's A tile.
__device__ __forceinline__ void gates_phase(const float* gi, const float* gh,
                                            const float* bih, const float* bhh,
                                            float* hold, _Float16* xOut, int tid)
{
  for (int idx = tid; idx < MBLK * KP; idx += 256) {
    const int m = idx >> 7, j = idx & 127;
    float hnew = 0.0f;
    if (j < HD) {
      float ir  = gi[m * NP3 + j]          + bih[j];
      float hr  = gh[m * NP3 + j]          + bhh[j];
      float iz  = gi[m * NP3 + HD + j]     + bih[HD + j];
      float hz  = gh[m * NP3 + HD + j]     + bhh[HD + j];
      float in_ = gi[m * NP3 + 2 * HD + j] + bih[2 * HD + j];
      float hn  = gh[m * NP3 + 2 * HD + j] + bhh[2 * HD + j];
      float r = sigmoidf_(ir + hr);
      float z = sigmoidf_(iz + hz);
      float nv = tanhf(in_ + r * hn);
      hnew = (1.0f - z) * nv + z * hold[m * NP1 + j];
      hold[m * NP1 + j] = hnew;
    }
    if (xOut) xOut[idx] = (_Float16)hnew;   // zero in pad region
  }
}

// ---------------------------------------------------------------------------
// Persistent fused 2-layer GRU: each block owns MBLK nodes for all 30 steps.
// Hidden state never leaves LDS until the end. ~120 KB LDS / block.
// ---------------------------------------------------------------------------
__global__ void __launch_bounds__(256)
gru_fused(const int* __restrict__ feat, const float* __restrict__ emb,
          const float* __restrict__ h0in,
          const _Float16* __restrict__ wih0, const _Float16* __restrict__ whh0,
          const _Float16* __restrict__ wih1, const _Float16* __restrict__ whh1,
          const float* __restrict__ bih0, const float* __restrict__ bhh0,
          const float* __restrict__ bih1, const float* __restrict__ bhh1,
          float* __restrict__ h1s)
{
  __shared__ alignas(32) _Float16 xA[MBLK * KP];
  __shared__ alignas(32) _Float16 hA[MBLK * KP];
  __shared__ float gi[MBLK * NP3];
  __shared__ float gh[MBLK * NP3];
  __shared__ float hold0[MBLK * NP1];
  __shared__ float hold1[MBLK * NP1];

  const int tid  = threadIdx.x;
  const int m0   = blockIdx.x * MBLK;
  const int wave = tid >> 5, lane = tid & 31;

  for (int idx = tid; idx < MBLK * NP1; idx += 256) {
    const int m = idx / NP1, j = idx % NP1;
    float v0 = 0.0f, v1 = 0.0f;
    if (j < HD) {
      v0 = h0in[(size_t)(m0 + m) * HD + j];
      v1 = h0in[(size_t)NN * HD + (size_t)(m0 + m) * HD + j];
    }
    hold0[idx] = v0; hold1[idx] = v1;
  }
  __syncthreads();

  for (int t = 0; t < TSEQ; ++t) {
    // A tiles: embedding gather (layer-0 x) and f16 copy of h0
    for (int idx = tid; idx < MBLK * KP; idx += 256) {
      const int m = idx >> 7, k = idx & 127;
      float xv = 0.0f;
      if (k < DIN) {
        const int tok = feat[(size_t)(m0 + m) * TSEQ + t];
        xv = emb[(size_t)tok * DIN + k];
      }
      float hv = (k < HD) ? hold0[m * NP1 + k] : 0.0f;
      xA[idx] = (_Float16)xv;
      hA[idx] = (_Float16)hv;
    }
    __syncthreads();
    matmul_phase(xA, hA, wih0, whh0, gi, gh, wave, lane);
    __syncthreads();
    gates_phase(gi, gh, bih0, bhh0, hold0, xA, tid);   // xA <- f16(h0_new)
    __syncthreads();
    for (int idx = tid; idx < MBLK * KP; idx += 256) { // hA <- f16(h1)
      const int m = idx >> 7, k = idx & 127;
      float hv = (k < HD) ? hold1[m * NP1 + k] : 0.0f;
      hA[idx] = (_Float16)hv;
    }
    __syncthreads();
    matmul_phase(xA, hA, wih1, whh1, gi, gh, wave, lane);
    __syncthreads();
    gates_phase(gi, gh, bih1, bhh1, hold1, nullptr, tid);
    __syncthreads();
  }

  for (int idx = tid; idx < MBLK * HD; idx += 256) {
    const int m = idx / HD, j = idx % HD;
    h1s[(size_t)(m0 + m) * HD + j] = hold1[m * NP1 + j];
  }
}

// ---------------------------------------------------------------------------
// Node-GEMM for the GCN convs: out[n,j] = sum_k X[n,k] * Wt[j,k]
// KPV is compile-time so B fragments stay in registers (no scratch spill).
// ---------------------------------------------------------------------------
template <int KPV>
__global__ void __launch_bounds__(256)
gemm_nodes(const float* __restrict__ X, int kvalid,
           const _Float16* __restrict__ Wt, int ntiles,
           float* __restrict__ out, int nvalid)
{
  constexpr int KCH = KPV / 32;
  __shared__ alignas(32) _Float16 Abuf[MBLK * KPV];
  const int m0 = blockIdx.x * MBLK;
  for (int idx = threadIdx.x; idx < MBLK * KPV; idx += 256) {
    const int m = idx / KPV, k = idx % KPV;
    float v = (k < kvalid) ? X[(size_t)(m0 + m) * kvalid + k] : 0.0f;
    Abuf[m * KPV + k] = (_Float16)v;
  }
  __syncthreads();
  const int wave = threadIdx.x >> 5, lane = threadIdx.x & 31;
  for (int nt = wave; nt < ntiles; nt += 8) {
    const _Float16* bp = Wt + (size_t)(nt * 16 + (lane & 15)) * KPV + ((lane >> 4) << 4);
    v16h breg[KCH];
#pragma unroll
    for (int kc = 0; kc < KCH; ++kc) breg[kc] = *(const v16h*)(bp + kc * 32);

    const int row0 = (lane >> 4) << 3;
    const int col  = nt * 16 + (lane & 15);
#pragma unroll
    for (int mt = 0; mt < MTILES; ++mt) {
      const _Float16* ap = Abuf + (size_t)(mt * 16 + (lane & 15)) * KPV + ((lane >> 4) << 4);
      v8f c = {};
#pragma unroll
      for (int kc = 0; kc < KCH; ++kc) {
        v16h a = *(const v16h*)(ap + kc * 32);
        c = wmma_f16(a, breg[kc], c);
      }
      if (col < nvalid) {
#pragma unroll
        for (int i = 0; i < 8; ++i)
          out[(size_t)(m0 + mt * 16 + row0 + i) * nvalid + col] = c[i];
      }
    }
  }
}

// ----------------------------- small kernels -------------------------------
__global__ void k_fill(float* p, float v, int n) {
  int i = blockIdx.x * 256 + threadIdx.x; if (i < n) p[i] = v;
}
// [300][100] fp32 -> [304][128] f16 (row-major, zero padded)
__global__ void k_convW3(const float* __restrict__ W, _Float16* __restrict__ out) {
  int i = blockIdx.x * 256 + threadIdx.x; if (i >= NP3 * KP) return;
  int j = i / KP, k = i % KP;
  out[i] = (_Float16)((j < 3 * HD && k < HD) ? W[j * HD + k] : 0.0f);
}
// W[K][Ncols] fp32 -> out[j][k] = W[k][j], rows padded to 112, cols to kp
__global__ void k_convWT(const float* __restrict__ W, _Float16* __restrict__ out,
                         int K, int Ncols, int kp) {
  int i = blockIdx.x * 256 + threadIdx.x; if (i >= NP1 * kp) return;
  int j = i / kp, k = i % kp;
  out[i] = (_Float16)((j < Ncols && k < K) ? W[k * Ncols + j] : 0.0f);
}
__global__ void k_deg(const int* __restrict__ ei, float* deg) {
  int e = blockIdx.x * 256 + threadIdx.x; if (e < NE) atomicAdd(&deg[ei[NE + e]], 1.0f);
}
__global__ void k_rsqrt(float* deg) {
  int i = blockIdx.x * 256 + threadIdx.x; if (i < NN) deg[i] = rsqrtf(deg[i]);
}
__global__ void k_scatter(const int* __restrict__ ei, const float* __restrict__ dinv,
                          const float* __restrict__ hbuf, float* __restrict__ out) {
  int idx = blockIdx.x * 256 + threadIdx.x; if (idx >= NE * HD) return;
  int e = idx / HD, h = idx % HD;
  int s = ei[e], d = ei[NE + e];
  atomicAdd(&out[(size_t)d * HD + h], hbuf[(size_t)s * HD + h] * dinv[s] * dinv[d]);
}
__global__ void k_selfbias(const float* __restrict__ hbuf, const float* __restrict__ dinv,
                           const float* __restrict__ b, float* __restrict__ out) {
  int idx = blockIdx.x * 256 + threadIdx.x; if (idx >= NN * HD) return;
  int n = idx / HD, h = idx % HD;
  out[idx] += hbuf[idx] * dinv[n] * dinv[n] + b[h];
}
__global__ void k_xcat(const float* __restrict__ conv1, const float* __restrict__ x1,
                       const int* __restrict__ indices, float* __restrict__ xcat) {
  int idx = blockIdx.x * 256 + threadIdx.x; if (idx >= NN * 2 * HD) return;
  int n = idx / (2 * HD), c = idx % (2 * HD);
  float v = (c < HD) ? conv1[(size_t)n * HD + c]
                     : x1[(size_t)indices[n] * HD + (c - HD)];
  xcat[idx] = eluf_(v);
}
__global__ void k_reduce(const float* __restrict__ conv2, const float* __restrict__ conv1raw,
                         const int* __restrict__ indices, float* tot, float* cnt) {
  int idx = blockIdx.x * 256 + threadIdx.x; if (idx >= NN * 2 * HD) return;
  int n = idx / (2 * HD), c = idx % (2 * HD);
  int b = indices[n];
  float v;
  if (c < HD) v = eluf_(conv2[(size_t)n * HD + c]);
  else        v = conv1raw[(size_t)indices[n] * HD + (c - HD)];
  atomicAdd(&tot[(size_t)b * 2 * HD + c], v);
  if (c == 0) atomicAdd(&cnt[b], 1.0f);
}
__global__ void k_final(const float* tot, const float* cnt, float* out) {
  int i = blockIdx.x * 256 + threadIdx.x; if (i >= BATCH * 2 * HD) return;
  float cn = cnt[i / (2 * HD)]; if (cn < 1.0f) cn = 1.0f;
  out[i] = tot[i] / cn;
}

// ---------------------------------------------------------------------------
extern "C" void kernel_launch(void* const* d_in, const int* in_sizes, int n_in,
                              void* d_out, int out_size, void* d_ws, size_t ws_size,
                              hipStream_t stream)
{
  const int*   feat    = (const int*)d_in[0];
  const int*   ei      = (const int*)d_in[1];
  const int*   indices = (const int*)d_in[2];
  const float* h0      = (const float*)d_in[3];
  const float* emb     = (const float*)d_in[4];
  const float* Wih0    = (const float*)d_in[5];
  const float* Whh0    = (const float*)d_in[6];
  const float* bih0    = (const float*)d_in[7];
  const float* bhh0    = (const float*)d_in[8];
  const float* Wih1    = (const float*)d_in[9];
  const float* Whh1    = (const float*)d_in[10];
  const float* bih1    = (const float*)d_in[11];
  const float* bhh1    = (const float*)d_in[12];
  const float* W1      = (const float*)d_in[13];
  const float* b1      = (const float*)d_in[14];
  const float* W2      = (const float*)d_in[15];
  const float* b2      = (const float*)d_in[16];
  float* out = (float*)d_out;

  char* ws = (char*)d_ws;
  size_t off = 0;
  auto alloc = [&](size_t bytes) -> void* {
    void* p = ws + off;
    off = (off + bytes + 255) & ~(size_t)255;
    return p;
  };
  float* h0s   = (float*)alloc((size_t)NN * HD * 4);      // reused as GEMM scratch (hbuf)
  float* h1s   = (float*)alloc((size_t)NN * HD * 4);      // hn = GCN input (x1)
  float* conv1 = (float*)alloc((size_t)NN * HD * 4);      // x2 (raw conv1 out)
  float* conv2 = (float*)alloc((size_t)NN * HD * 4);
  float* xcat  = (float*)alloc((size_t)NN * 2 * HD * 4);
  float* deg   = (float*)alloc((size_t)NN * 4);           // becomes dinv
  float* tot   = (float*)alloc((size_t)BATCH * 2 * HD * 4);
  float* cnt   = (float*)alloc((size_t)BATCH * 4);
  _Float16* wih0h = (_Float16*)alloc((size_t)NP3 * KP * 2);
  _Float16* whh0h = (_Float16*)alloc((size_t)NP3 * KP * 2);
  _Float16* wih1h = (_Float16*)alloc((size_t)NP3 * KP * 2);
  _Float16* whh1h = (_Float16*)alloc((size_t)NP3 * KP * 2);
  _Float16* w1h   = (_Float16*)alloc((size_t)NP1 * KP * 2);
  _Float16* w2h   = (_Float16*)alloc((size_t)NP1 * KP2 * 2);

  // --- weight prep (f16, padded, WMMA-friendly layouts) ---
  k_convW3<<<(NP3 * KP + 255) / 256, 256, 0, stream>>>(Wih0, wih0h);
  k_convW3<<<(NP3 * KP + 255) / 256, 256, 0, stream>>>(Whh0, whh0h);
  k_convW3<<<(NP3 * KP + 255) / 256, 256, 0, stream>>>(Wih1, wih1h);
  k_convW3<<<(NP3 * KP + 255) / 256, 256, 0, stream>>>(Whh1, whh1h);
  k_convWT<<<(NP1 * KP + 255) / 256, 256, 0, stream>>>(W1, w1h, HD, HD, KP);
  k_convWT<<<(NP1 * KP2 + 255) / 256, 256, 0, stream>>>(W2, w2h, 2 * HD, HD, KP2);

  // --- fused 2-layer GRU (persistent per 32-node tile, WMMA) ---
  gru_fused<<<NN / MBLK, 256, 0, stream>>>(feat, emb, h0,
                                           wih0h, whh0h, wih1h, whh1h,
                                           bih0, bhh0, bih1, bhh1, h1s);

  // --- degrees with self-loops, dinv ---
  k_fill<<<(NN + 255) / 256, 256, 0, stream>>>(deg, 1.0f, NN);
  k_deg<<<(NE + 255) / 256, 256, 0, stream>>>(ei, deg);
  k_rsqrt<<<(NN + 255) / 256, 256, 0, stream>>>(deg);

  // --- GCNConv 1: h = hn @ W1 ; aggregate ; + self-loop + b1 ---
  gemm_nodes<KP><<<NN / MBLK, 256, 0, stream>>>(h1s, HD, w1h, NT1, h0s, HD);
  k_fill<<<(NN * HD + 255) / 256, 256, 0, stream>>>(conv1, 0.0f, NN * HD);
  k_scatter<<<(NE * HD + 255) / 256, 256, 0, stream>>>(ei, deg, h0s, conv1);
  k_selfbias<<<(NN * HD + 255) / 256, 256, 0, stream>>>(h0s, deg, b1, conv1);
  k_xcat<<<(NN * 2 * HD + 255) / 256, 256, 0, stream>>>(conv1, h1s, indices, xcat);

  // --- GCNConv 2: h = elu(cat) @ W2 ; aggregate ; + self-loop + b2 ---
  gemm_nodes<KP2><<<NN / MBLK, 256, 0, stream>>>(xcat, 2 * HD, w2h, NT1, h0s, HD);
  k_fill<<<(NN * HD + 255) / 256, 256, 0, stream>>>(conv2, 0.0f, NN * HD);
  k_scatter<<<(NE * HD + 255) / 256, 256, 0, stream>>>(ei, deg, h0s, conv2);
  k_selfbias<<<(NN * HD + 255) / 256, 256, 0, stream>>>(h0s, deg, b2, conv2);

  // --- elu / concat / scatter-mean ---
  k_fill<<<(BATCH * 2 * HD + 255) / 256, 256, 0, stream>>>(tot, 0.0f, BATCH * 2 * HD);
  k_fill<<<1, 256, 0, stream>>>(cnt, 0.0f, BATCH);
  k_reduce<<<(NN * 2 * HD + 255) / 256, 256, 0, stream>>>(conv2, conv1, indices, tot, cnt);
  k_final<<<(BATCH * 2 * HD + 255) / 256, 256, 0, stream>>>(tot, cnt, out);
}
